// SBNLinear_39754217292616
// MI455X (gfx1250) — matrix operand order repaired and checked by reading, not compiled, not run on hardware
//
#include <hip/hip_runtime.h>
#include <hip/hip_bf16.h>

typedef __attribute__((ext_vector_type(16))) _Float16 v16h;
typedef __attribute__((ext_vector_type(8)))  _Float16 v8h;
typedef __attribute__((ext_vector_type(8)))  float    v8f;
typedef __attribute__((ext_vector_type(4)))  float    v4f;

namespace {
constexpr int M_DIM = 1024;   // tokens
constexpr int N_DIM = 4096;   // out features
constexpr int K_DIM = 4096;   // in features
constexpr int BM = 128;
constexpr int BN = 128;
constexpr int BK = 32;        // one WMMA K-step
constexpr int LDSP = BK + 8;  // padded LDS row stride in f16 (80 B, keeps 16B alignment)
}

union FragU  { v16h f; v8h h[2]; };
union Pack16 { v8h v[2]; _Float16 e[16]; };

// A fragment (16x32 f16, MxK): lane l -> row M=l&15; sel=l>>4.
// v16h element j: j<8 -> K = sel*8 + j ; j>=8 -> K = 16 + sel*8 + (j-8).
__device__ __forceinline__ v16h ld_fragA(const _Float16* row_base) {
  FragU f;
  f.h[0] = *(const v8h*)(row_base);        // K = sel*8 + 0..7
  f.h[1] = *(const v8h*)(row_base + 16);   // K = 16 + sel*8 + 0..7
  return f.f;
}

// B fragment (32x16 f16, KxN): lane l -> col N=l&15; sel=l>>4.
// v16h element j: j<8 -> K = sel*8 + j ; j>=8 -> K = 16 + sel*8 + (j-8).
__device__ __forceinline__ v16h ld_fragB(const _Float16* row_base) {
  FragU f;
  f.h[0] = *(const v8h*)(row_base);        // K = sel*8 + 0..7
  f.h[1] = *(const v8h*)(row_base + 16);   // K = 16 + sel*8 + 0..7
  return f.f;
}

__global__ __launch_bounds__(256)
void sbn_wmma_gemm(const float* __restrict__ X,   // [M,K] f32
                   const float* __restrict__ W,   // [N,K] f32
                   const float* __restrict__ S,   // [N]   f32
                   float* __restrict__ Out)       // [M,N] f32
{
  __shared__ _Float16 sAhi[BM * LDSP];
  __shared__ _Float16 sAlo[BM * LDSP];
  __shared__ _Float16 sB  [BN * LDSP];

  const int tid   = threadIdx.x;
  const int lane  = tid & 31;
  const int wave  = tid >> 5;
  const int waveM = wave & 3;   // 0..3 -> 4 x 32 rows
  const int waveN = wave >> 2;  // 0..1 -> 2 x 64 cols
  const int lrow  = lane & 15;
  const int lsel  = lane >> 4;

  const int blockN = blockIdx.x;
  const int blockM = blockIdx.y;

  // Global->LDS staging map: thread t covers row r = t>>1, 16 cols at h*16.
  const int r = tid >> 1;
  const int h = tid & 1;
  const float* gA = X + (size_t)(blockM * BM + r) * K_DIM + h * 16;
  const float* gB = W + (size_t)(blockN * BN + r) * K_DIM + h * 16;
  _Float16* pAhi = sAhi + r * LDSP + h * 16;
  _Float16* pAlo = sAlo + r * LDSP + h * 16;
  _Float16* pB   = sB   + r * LDSP + h * 16;

  v8f acc[2][4];
#pragma unroll
  for (int mf = 0; mf < 2; ++mf)
#pragma unroll
    for (int nf = 0; nf < 4; ++nf)
      acc[mf][nf] = (v8f){0.f, 0.f, 0.f, 0.f, 0.f, 0.f, 0.f, 0.f};

  for (int k0 = 0; k0 < K_DIM; k0 += BK) {
    // Issue next-tile global loads before the barrier (overlaps prior compute).
    v4f a[4], b[4];
#pragma unroll
    for (int i = 0; i < 4; ++i) a[i] = *(const v4f*)(gA + k0 + i * 4);
#pragma unroll
    for (int i = 0; i < 4; ++i) b[i] = *(const v4f*)(gB + k0 + i * 4);

    const int kp = k0 + 4 * BK;
    if (kp < K_DIM) {                       // uniform branch
      __builtin_prefetch(gA + kp, 0, 3);    // -> global_prefetch_b8
      __builtin_prefetch(gB + kp, 0, 3);
    }

    __syncthreads();  // previous tile fully consumed

    // Convert in registers, stage to LDS.
    Pack16 th, tl, tb;
#pragma unroll
    for (int i = 0; i < 4; ++i) {
#pragma unroll
      for (int j = 0; j < 4; ++j) {
        const float v  = a[i][j];
        const _Float16 hi = (_Float16)v;
        th.e[i * 4 + j] = hi;
        tl.e[i * 4 + j] = (_Float16)(v - (float)hi);  // residual term
        tb.e[i * 4 + j] = (b[i][j] > 0.f) ? (_Float16)1.0f : (_Float16)0.0f;
      }
    }
    *(v8h*)(pAhi)     = th.v[0];  *(v8h*)(pAhi + 8) = th.v[1];
    *(v8h*)(pAlo)     = tl.v[0];  *(v8h*)(pAlo + 8) = tl.v[1];
    *(v8h*)(pB)       = tb.v[0];  *(v8h*)(pB   + 8) = tb.v[1];

    __syncthreads();  // tile visible to all waves

    // B fragments for this wave's 64-column strip.
    v16h bf[4];
#pragma unroll
    for (int nf = 0; nf < 4; ++nf)
      bf[nf] = ld_fragB(sB + (waveN * 64 + nf * 16 + lrow) * LDSP + lsel * 8);

#pragma unroll
    for (int mf = 0; mf < 2; ++mf) {
      const int arow = (waveM * 32 + mf * 16 + lrow) * LDSP + lsel * 8;
      const v16h ahi = ld_fragA(sAhi + arow);
      const v16h alo = ld_fragA(sAlo + arow);
#pragma unroll
      for (int nf = 0; nf < 4; ++nf) {
        acc[mf][nf] = __builtin_amdgcn_wmma_f32_16x16x32_f16(
            false, ahi, false, bf[nf], (short)0, acc[mf][nf], false, false);
        acc[mf][nf] = __builtin_amdgcn_wmma_f32_16x16x32_f16(
            false, alo, false, bf[nf], (short)0, acc[mf][nf], false, false);
      }
    }
  }

  // Epilogue: scale by rint(max(s,1)) (ties-to-even, matches jnp.round) and store.
  // C layout: lane l -> N = l&15 ; VGPR v -> M = v + (l>=16 ? 8 : 0).
#pragma unroll
  for (int nf = 0; nf < 4; ++nf) {
    const int n = blockN * BN + waveN * 64 + nf * 16 + lrow;
    const float se = rintf(fmaxf(S[n], 1.0f));
#pragma unroll
    for (int mf = 0; mf < 2; ++mf) {
      const int m0 = blockM * BM + waveM * 32 + mf * 16 + (lsel << 3);
#pragma unroll
      for (int v = 0; v < 8; ++v) {
        Out[(size_t)(m0 + v) * N_DIM + n] = acc[mf][nf][v] * se;
      }
    }
  }
}

extern "C" void kernel_launch(void* const* d_in, const int* in_sizes, int n_in,
                              void* d_out, int out_size, void* d_ws, size_t ws_size,
                              hipStream_t stream) {
  (void)in_sizes; (void)n_in; (void)d_ws; (void)ws_size; (void)out_size;
  const float* X = (const float*)d_in[0];  // x [1024,4096]
  const float* W = (const float*)d_in[1];  // weight [4096,4096]
  const float* S = (const float*)d_in[2];  // scaling_factor [4096]
  float* Out = (float*)d_out;              // [1024,4096]

  dim3 grid(N_DIM / BN, M_DIM / BM);       // (32, 8)
  sbn_wmma_gemm<<<grid, dim3(256), 0, stream>>>(X, W, S, Out);
}
